// MixtureOfExperts_32006096290575
// MI455X (gfx1250) — matrix operand order repaired and checked by reading, not compiled, not run on hardware
//
#include <hip/hip_runtime.h>

// ---------------- problem constants ----------------
#define B_NUM   4
#define T_NUM   2048
#define D_DIM   1024
#define F_DIM   4096
#define E_NUM   8
#define NTOK    (B_NUM * T_NUM)        // 8192
#define ALPHA_F 0.01f

// GEMM tiling
#define BM 128
#define BN 64
#define KSTEP 64
#define LDPAD 72                        // LDS row stride in elements (144B, 16B-aligned)

typedef __attribute__((ext_vector_type(16))) __bf16 v16bf;
typedef __attribute__((ext_vector_type(8)))  float  v8f;

union FragAB { v16bf v; unsigned u[8]; uint4 q[2]; };

// ---------------- fp32 -> bf16 (packed hardware cvt when available) ----------------
#if __has_builtin(__builtin_amdgcn_cvt_pk_bf16_f32)
typedef __attribute__((ext_vector_type(2))) __bf16 v2bf;
__device__ __forceinline__ unsigned pkbf(float a, float b) {
    union { v2bf v; unsigned u; } c;
    c.v = __builtin_amdgcn_cvt_pk_bf16_f32(a, b);   // v_cvt_pk_bf16_f32
    return c.u;
}
__device__ __forceinline__ unsigned short f2bf(float f) {
    return (unsigned short)(pkbf(f, f) & 0xffffu);
}
#else
__device__ __forceinline__ unsigned short f2bf(float f) {
    union { float f; unsigned u; } v; v.f = f;
    unsigned r = v.u + 0x7FFFu + ((v.u >> 16) & 1u);   // round-to-nearest-even
    return (unsigned short)(r >> 16);
}
__device__ __forceinline__ unsigned pkbf(float a, float b) {
    return (unsigned)f2bf(a) | ((unsigned)f2bf(b) << 16);
}
#endif

// ---------------- async global->LDS copy (CDNA5 path when available) ----------------
#if __has_builtin(__builtin_amdgcn_global_load_async_to_lds_b128)
#define HAVE_ASYNC_LDS 1
typedef int async_b128_t __attribute__((vector_size(16)));   // matches builtin param type
#define GAS(p) ((async_b128_t*)(p))
#define LAS(p) ((async_b128_t*)(p))
__device__ __forceinline__ void async_wait0() {
#if __has_builtin(__builtin_amdgcn_s_wait_asynccnt)
    __builtin_amdgcn_s_wait_asynccnt(0);
#else
    asm volatile("s_wait_asynccnt 0x0" ::: "memory");
#endif
}
#else
#define HAVE_ASYNC_LDS 0
#endif

// ---------------- kernel 1: zero output + counters ----------------
__global__ void moe_init(float* __restrict__ out, int n, int* __restrict__ counts) {
    int stride = gridDim.x * blockDim.x;
    for (int i = blockIdx.x * blockDim.x + threadIdx.x; i < n; i += stride)
        out[i] = 0.0f;
    if (blockIdx.x == 0 && threadIdx.x < E_NUM)
        counts[threadIdx.x] = 0;
}

// ---------------- kernel 2: x fp32 -> bf16 ----------------
__global__ void moe_xconv(const float* __restrict__ x, unsigned short* __restrict__ xb) {
    int i = (blockIdx.x * blockDim.x + threadIdx.x) * 4;
    float4 v = *(const float4*)(x + i);
    uint2 o;
    o.x = pkbf(v.x, v.y);
    o.y = pkbf(v.z, v.w);
    *(uint2*)(xb + i) = o;
}

// ---------------- kernel 3: router ----------------
// 1 wave per token; router_w staged in LDS (32 KB)
__global__ void moe_router(const float* __restrict__ x,
                           const float* __restrict__ rw,
                           float* __restrict__ Pbuf,          // [NTOK][E]
                           int*   __restrict__ counts,        // [E]
                           int*   __restrict__ lists,         // [E][NTOK]
                           float* __restrict__ wlists) {      // [E][NTOK]
    __shared__ float rws[E_NUM * D_DIM];
    for (int i = threadIdx.x; i < E_NUM * D_DIM; i += blockDim.x)
        rws[i] = rw[i];
    __syncthreads();

    int wave = threadIdx.x >> 5, lane = threadIdx.x & 31;
    int token = blockIdx.x * 8 + wave;

    float acc[E_NUM];
#pragma unroll
    for (int e = 0; e < E_NUM; ++e) acc[e] = 0.0f;

    const float* xrow = x + (size_t)token * D_DIM;
    for (int d = lane; d < D_DIM; d += 32) {
        float xv = xrow[d];
#pragma unroll
        for (int e = 0; e < E_NUM; ++e) acc[e] += xv * rws[e * D_DIM + d];
    }
#pragma unroll
    for (int e = 0; e < E_NUM; ++e)
#pragma unroll
        for (int off = 16; off > 0; off >>= 1)
            acc[e] += __shfl_xor(acc[e], off, 32);

    if (lane == 0) {
        float m = acc[0];
#pragma unroll
        for (int e = 1; e < E_NUM; ++e) m = fmaxf(m, acc[e]);
        float p[E_NUM], s = 0.0f;
#pragma unroll
        for (int e = 0; e < E_NUM; ++e) { p[e] = __expf(acc[e] - m); s += p[e]; }
        float inv = __builtin_amdgcn_rcpf(s);
#pragma unroll
        for (int e = 0; e < E_NUM; ++e) Pbuf[(size_t)token * E_NUM + e] = p[e] * inv;

        int e0 = 0;
#pragma unroll
        for (int e = 1; e < E_NUM; ++e) if (acc[e] > acc[e0]) e0 = e;
        int e1 = (e0 == 0) ? 1 : 0;
#pragma unroll
        for (int e = 0; e < E_NUM; ++e)
            if (e != e0 && acc[e] > acc[e1]) e1 = e;

        float w0 = __builtin_amdgcn_rcpf(1.0f + __expf(acc[e1] - acc[e0])); // top-2 softmax
        float w1 = 1.0f - w0;

        int p0 = atomicAdd(&counts[e0], 1);
        lists[e0 * NTOK + p0]  = token;
        wlists[e0 * NTOK + p0] = w0;
        int p1 = atomicAdd(&counts[e1], 1);
        lists[e1 * NTOK + p1]  = token;
        wlists[e1 * NTOK + p1] = w1;
    }
}

// ---------------- kernel 4: aux loss (deterministic reduction) ----------------
__global__ void moe_aux(const int* __restrict__ counts,
                        const float* __restrict__ Pbuf,
                        float* __restrict__ out) {
    __shared__ float red[256][E_NUM];
    float loc[E_NUM];
#pragma unroll
    for (int e = 0; e < E_NUM; ++e) loc[e] = 0.0f;
    for (int t = threadIdx.x; t < NTOK; t += 256)
#pragma unroll
        for (int e = 0; e < E_NUM; ++e) loc[e] += Pbuf[(size_t)t * E_NUM + e];
#pragma unroll
    for (int e = 0; e < E_NUM; ++e) red[threadIdx.x][e] = loc[e];
    __syncthreads();
    for (int s = 128; s > 0; s >>= 1) {
        if (threadIdx.x < s)
#pragma unroll
            for (int e = 0; e < E_NUM; ++e) red[threadIdx.x][e] += red[threadIdx.x + s][e];
        __syncthreads();
    }
    if (threadIdx.x == 0) {
        float aux = 0.0f;
#pragma unroll
        for (int e = 0; e < E_NUM; ++e) {
            float f = (float)counts[e] / (float)NTOK;
            float P = red[0][e] / (float)NTOK;
            aux += f * P;
        }
        out[(size_t)NTOK * D_DIM] = ALPHA_F * (float)E_NUM * aux;
    }
}

// ---------------- WMMA fragment helpers (per ISA 7.12.2 layouts) ----------------
// A 16x32 bf16: lane<16 -> M=lane, K in {0..7,16..23}; lane>=16 -> M=lane-16, K in {8..15,24..31}
__device__ __forceinline__ void load_fragA(FragAB& f, const unsigned short* __restrict__ base,
                                           int m, int ks, int lane) {
    const unsigned short* row = base + m * LDPAD + ks;
    int kb = (lane < 16) ? 0 : 8;
#pragma unroll
    for (int j = 0; j < 4; ++j) {
        f.u[j]     = *(const unsigned*)(row + kb + 2 * j);
        f.u[4 + j] = *(const unsigned*)(row + 16 + kb + 2 * j);
    }
}
// B 32x16 bf16 (LDS stored [N][K]): lane<16 -> N=lane, K in {0..7,16..23} ; lane>=16 -> N=lane-16, K in {8..15,24..31}
__device__ __forceinline__ void load_fragB(FragAB& f, const unsigned short* __restrict__ base,
                                           int n, int ks, int lane) {
    const unsigned short* row = base + n * LDPAD + ks + ((lane < 16) ? 0 : 8);
    const uint4* p = (const uint4*)row;
    f.q[0] = p[0];
    f.q[1] = p[2];
}

// transpose-stage a 64(K) x 64(N) fp32 tile as bf16 LDS [N][K], K-pairs packed as b32 stores.
// thread t handles a 4K x 4N sub-block; global reads coalesce along N.
#define STP(buf, n, k, v) (*(unsigned*)((buf) + (n) * LDPAD + (k)) = (v))
__device__ __forceinline__ void stage_wtile(unsigned short* __restrict__ dst,
                                            const float* __restrict__ src, // &w[kk][n0]
                                            long ldn, int tid, bool pf) {
    int kp = (tid >> 4) * 4;       // 0,4,...,60
    int nq = (tid & 15) * 4;       // 0,4,...,60
    const float* p = src + (size_t)kp * ldn + nq;
    float4 r0 = *(const float4*)(p);
    float4 r1 = *(const float4*)(p + ldn);
    float4 r2 = *(const float4*)(p + 2 * ldn);
    float4 r3 = *(const float4*)(p + 3 * ldn);
    if (pf) __builtin_prefetch(p + (size_t)KSTEP * ldn, 0, 1);  // next K-chunk
    STP(dst, nq + 0, kp,     pkbf(r0.x, r1.x)); STP(dst, nq + 0, kp + 2, pkbf(r2.x, r3.x));
    STP(dst, nq + 1, kp,     pkbf(r0.y, r1.y)); STP(dst, nq + 1, kp + 2, pkbf(r2.y, r3.y));
    STP(dst, nq + 2, kp,     pkbf(r0.z, r1.z)); STP(dst, nq + 2, kp + 2, pkbf(r2.z, r3.z));
    STP(dst, nq + 3, kp,     pkbf(r0.w, r1.w)); STP(dst, nq + 3, kp + 2, pkbf(r2.w, r3.w));
}

// copy one 128-row x 64-col bf16 tile global -> LDS (async path when available).
// thread t covers 32 elements (64B) of one row half.
__device__ __forceinline__ void stage_atile(unsigned short* __restrict__ lds,
                                            const unsigned short* __restrict__ grow,
                                            int tid) {
    int m = tid >> 1, half = tid & 1;
    const unsigned short* gsrc = grow;          // caller passes &row[m][kk] base per thread
    unsigned short* ldst = lds + m * LDPAD + half * 32;
#if HAVE_ASYNC_LDS
#pragma unroll
    for (int q = 0; q < 4; ++q)
        __builtin_amdgcn_global_load_async_to_lds_b128(
            GAS(gsrc + q * 8), LAS(ldst + q * 8), 0, 0);
#else
    const uint4* s4 = (const uint4*)gsrc;
    uint4 a = s4[0], b = s4[1], c = s4[2], d = s4[3];
    uint4* d4 = (uint4*)ldst;
    d4[0] = a; d4[1] = b; d4[2] = c; d4[3] = d;
#endif
}

__device__ __forceinline__ void tile_fence() {
#if HAVE_ASYNC_LDS
    async_wait0();      // our async LDS writes complete before the barrier
#endif
    __syncthreads();
}

// ---------------- kernel 5: gate/up GEMMs + SiLU*u -> h (bf16) ----------------
__global__ __launch_bounds__(256)
void moe_gateup(const unsigned short* __restrict__ xb,
                const float* __restrict__ wg, const float* __restrict__ wu,
                const int* __restrict__ lists, const int* __restrict__ counts,
                unsigned short* __restrict__ h, int e) {
    const int cnt = counts[e];
    const int m0  = blockIdx.y * BM;
    if (m0 >= cnt) return;
    const int n0  = blockIdx.x * BN;

    __shared__ unsigned short Xs[BM * LDPAD];   // 18 KB
    __shared__ unsigned short Gs[BN * LDPAD];   //  9 KB
    __shared__ unsigned short Us[BN * LDPAD];   //  9 KB
    __shared__ int tokL[BM];

    const int tid = threadIdx.x, lane = tid & 31, wave = tid >> 5;
    const int wm = wave & 3, wn = wave >> 2;    // 4x2 wave grid, 32x32 per wave

    if (tid < BM) {
        int pos = m0 + tid;
        tokL[tid] = (pos < cnt) ? lists[e * NTOK + pos] : 0;
    }
    __syncthreads();

    const float* wgE = wg + (size_t)e * D_DIM * F_DIM;
    const float* wuE = wu + (size_t)e * D_DIM * F_DIM;
    const size_t xrow = (size_t)tokL[tid >> 1] * D_DIM + (tid & 1) * 32;

    v8f cg[2][2], cu[2][2];
#pragma unroll
    for (int i = 0; i < 2; ++i)
#pragma unroll
        for (int j = 0; j < 2; ++j) { cg[i][j] = (v8f)0.0f; cu[i][j] = (v8f)0.0f; }

    for (int kk = 0; kk < D_DIM; kk += KSTEP) {
        stage_atile(Xs, xb + xrow + kk, tid);   // gathered X rows, already bf16
        bool pf = (kk + KSTEP) < D_DIM;
        stage_wtile(Gs, wgE + (size_t)kk * F_DIM + n0, F_DIM, tid, pf);
        stage_wtile(Us, wuE + (size_t)kk * F_DIM + n0, F_DIM, tid, pf);
        tile_fence();

#pragma unroll
        for (int ks = 0; ks < KSTEP; ks += 32) {
            FragAB af[2], bg[2], bu[2];
#pragma unroll
            for (int mi = 0; mi < 2; ++mi)
                load_fragA(af[mi], Xs, wm * 32 + mi * 16 + (lane & 15), ks, lane);
#pragma unroll
            for (int ni = 0; ni < 2; ++ni) {
                load_fragB(bg[ni], Gs, wn * 32 + ni * 16 + (lane & 15), ks, lane);
                load_fragB(bu[ni], Us, wn * 32 + ni * 16 + (lane & 15), ks, lane);
            }
#pragma unroll
            for (int mi = 0; mi < 2; ++mi)
#pragma unroll
                for (int ni = 0; ni < 2; ++ni) {
                    cg[mi][ni] = __builtin_amdgcn_wmma_f32_16x16x32_bf16(
                        false, af[mi].v, false, bg[ni].v, (short)0, cg[mi][ni], false, false);
                    cu[mi][ni] = __builtin_amdgcn_wmma_f32_16x16x32_bf16(
                        false, af[mi].v, false, bu[ni].v, (short)0, cu[mi][ni], false, false);
                }
        }
        __syncthreads();
    }

    // epilogue: h = silu(g) * u   (C layout: VGPR r -> M=r (+8 for upper lanes), N=lane&15)
#pragma unroll
    for (int mi = 0; mi < 2; ++mi)
#pragma unroll
        for (int ni = 0; ni < 2; ++ni) {
            int Mb = m0 + wm * 32 + mi * 16 + ((lane < 16) ? 0 : 8);
            int Nc = n0 + wn * 32 + ni * 16 + (lane & 15);
#pragma unroll
            for (int r = 0; r < 8; ++r) {
                int pos = Mb + r;
                if (pos < cnt) {
                    float g = cg[mi][ni][r];
                    float u = cu[mi][ni][r];
                    float sig = __builtin_amdgcn_rcpf(1.0f + __expf(-g)); // v_rcp_f32
                    h[(size_t)pos * F_DIM + Nc] = f2bf(g * sig * u);
                }
            }
        }
}

// ---------------- kernel 6: down GEMM, scale by combine weight, += out ----------------
__global__ __launch_bounds__(256)
void moe_down(const unsigned short* __restrict__ h,
              const float* __restrict__ wd,
              const int* __restrict__ lists, const float* __restrict__ wlists,
              const int* __restrict__ counts,
              float* __restrict__ out, int e) {
    const int cnt = counts[e];
    const int m0  = blockIdx.y * BM;
    if (m0 >= cnt) return;
    const int n0  = blockIdx.x * BN;

    __shared__ unsigned short Hs[BM * LDPAD];
    __shared__ unsigned short Ws[BN * LDPAD];
    __shared__ int   tokL[BM];
    __shared__ float wtL[BM];

    const int tid = threadIdx.x, lane = tid & 31, wave = tid >> 5;
    const int wm = wave & 3, wn = wave >> 2;

    if (tid < BM) {
        int pos = m0 + tid;
        if (pos < cnt) { tokL[tid] = lists[e * NTOK + pos]; wtL[tid] = wlists[e * NTOK + pos]; }
        else           { tokL[tid] = 0;                     wtL[tid] = 0.0f; }
    }
    __syncthreads();

    const float* wdE = wd + (size_t)e * F_DIM * D_DIM;
    const size_t hrow = (size_t)(m0 + (tid >> 1)) * F_DIM + (tid & 1) * 32;

    v8f c[2][2];
#pragma unroll
    for (int i = 0; i < 2; ++i)
#pragma unroll
        for (int j = 0; j < 2; ++j) c[i][j] = (v8f)0.0f;

    for (int kk = 0; kk < F_DIM; kk += KSTEP) {
        stage_atile(Hs, h + hrow + kk, tid);    // contiguous h rows (list positions)
        stage_wtile(Ws, wdE + (size_t)kk * D_DIM + n0, D_DIM, tid, (kk + KSTEP) < F_DIM);
        tile_fence();

#pragma unroll
        for (int ks = 0; ks < KSTEP; ks += 32) {
            FragAB af[2], bf[2];
#pragma unroll
            for (int mi = 0; mi < 2; ++mi)
                load_fragA(af[mi], Hs, wm * 32 + mi * 16 + (lane & 15), ks, lane);
#pragma unroll
            for (int ni = 0; ni < 2; ++ni)
                load_fragB(bf[ni], Ws, wn * 32 + ni * 16 + (lane & 15), ks, lane);
#pragma unroll
            for (int mi = 0; mi < 2; ++mi)
#pragma unroll
                for (int ni = 0; ni < 2; ++ni)
                    c[mi][ni] = __builtin_amdgcn_wmma_f32_16x16x32_bf16(
                        false, af[mi].v, false, bf[ni].v, (short)0, c[mi][ni], false, false);
        }
        __syncthreads();
    }

#pragma unroll
    for (int mi = 0; mi < 2; ++mi)
#pragma unroll
        for (int ni = 0; ni < 2; ++ni) {
            int mloc = wm * 32 + mi * 16 + ((lane < 16) ? 0 : 8);
            int Nc   = n0 + wn * 32 + ni * 16 + (lane & 15);
#pragma unroll
            for (int r = 0; r < 8; ++r) {
                int pos = m0 + mloc + r;
                if (pos < cnt) {
                    int tok = tokL[mloc + r];
                    float w = wtL[mloc + r];
                    size_t o = (size_t)tok * D_DIM + Nc;
                    out[o] += w * c[mi][ni][r];   // one writer per element per expert pass
                }
            }
        }
}

// ---------------- launcher ----------------
extern "C" void kernel_launch(void* const* d_in, const int* in_sizes, int n_in,
                              void* d_out, int out_size, void* d_ws, size_t ws_size,
                              hipStream_t stream) {
    const float* x  = (const float*)d_in[0];
    const float* rw = (const float*)d_in[1];
    const float* wg = (const float*)d_in[2];
    const float* wu = (const float*)d_in[3];
    const float* wd = (const float*)d_in[4];
    float* out = (float*)d_out;

    char* ws = (char*)d_ws;
    size_t off = 0;
    auto carve = [&](size_t bytes) -> char* {
        char* p = ws + off;
        off = (off + bytes + 255) & ~(size_t)255;
        return p;
    };
    unsigned short* xb     = (unsigned short*)carve((size_t)NTOK * D_DIM * 2);   // 16 MB
    unsigned short* hbuf   = (unsigned short*)carve((size_t)NTOK * F_DIM * 2);   // 64 MB (one expert)
    int*            lists  = (int*)           carve((size_t)E_NUM * NTOK * 4);
    float*          wlists = (float*)         carve((size_t)E_NUM * NTOK * 4);
    int*            counts = (int*)           carve(64);
    float*          Pbuf   = (float*)         carve((size_t)NTOK * E_NUM * 4);

    const int nOut = NTOK * D_DIM + 1;

    moe_init  <<<2048, 256, 0, stream>>>(out, nOut, counts);
    moe_xconv <<<(NTOK * D_DIM) / 1024, 256, 0, stream>>>(x, xb);
    moe_router<<<NTOK / 8, 256, 0, stream>>>(x, rw, Pbuf, counts, lists, wlists);
    moe_aux   <<<1, 256, 0, stream>>>(counts, Pbuf, out);

    for (int e = 0; e < E_NUM; ++e) {
        moe_gateup<<<dim3(F_DIM / BN, NTOK / BM), 256, 0, stream>>>(
            xb, wg, wu, lists, counts, hbuf, e);
        moe_down  <<<dim3(D_DIM / BN, NTOK / BM), 256, 0, stream>>>(
            hbuf, wd, lists, wlists, counts, out, e);
    }
}
